// TopKNet_970662609216
// MI455X (gfx1250) — matrix operand order repaired and checked by reading, not compiled, not run on hardware
//
#include <hip/hip_runtime.h>
#include <hip/hip_bf16.h>

typedef __bf16 bf16_t;
typedef __attribute__((ext_vector_type(16))) __bf16 v16bf;
typedef __attribute__((ext_vector_type(8)))  __bf16 v8bf;
typedef __attribute__((ext_vector_type(8)))  float   v8f;

#define B_G   32
#define N0    1024
#define E_G   8192
#define NE    (B_G * E_G)     // 262144 edges
#define F_IN  128
#define HDIM  256
#define NCLS  10
#define KP1   820
#define KP2   656
#define KP3   525
#define NODES0 (B_G * N0)     // 32768

// ---------- helpers ----------
__device__ __forceinline__ unsigned f2ord(float f) {
    unsigned u = __float_as_uint(f);
    return (u & 0x80000000u) ? ~u : (u | 0x80000000u);
}
__device__ __forceinline__ float ord2f(unsigned u) {
    unsigned b = (u & 0x80000000u) ? (u & 0x7fffffffu) : ~u;
    return __uint_as_float(b);
}
__device__ __forceinline__ float wave_sum(float v) {
    for (int off = 16; off > 0; off >>= 1) v += __shfl_down(v, off, 32);
    return v;
}

// ---------- fills ----------
__global__ void fill_f32(float* p, float v, long n) {
    long i = (long)blockIdx.x * blockDim.x + threadIdx.x;
    if (i < n) p[i] = v;
}
__global__ void fill_i32(int* p, int v, long n) {
    long i = (long)blockIdx.x * blockDim.x + threadIdx.x;
    if (i < n) p[i] = v;
}
__global__ void init_edges(const int* __restrict__ s, const int* __restrict__ d,
                           int* ns, int* nd, int* nm, int ne) {
    int e = blockIdx.x * blockDim.x + threadIdx.x;
    if (e < ne) { ns[e] = s[e]; nd[e] = d[e]; nm[e] = 1; }
}

// ---------- fp32 -> bf16 (optional fused eval-BN on input) ----------
__global__ void cvt_bf16(const float* __restrict__ x, bf16_t* __restrict__ y, long n, int F,
                         const float* g, const float* b, const float* m, const float* v) {
    long i = (long)blockIdx.x * blockDim.x + threadIdx.x;
    if (i >= n) return;
    float val = x[i];
    if (g) {
        int f = (int)(i % F);
        val = (val - m[f]) * rsqrtf(v[f] + 1e-5f) * g[f] + b[f];
    }
    y[i] = (bf16_t)val;
}

// ---------- pack W[K,N] fp32 row-major into per-lane WMMA B fragments (bf16) ----------
// fragment element e (0..15) of lane L, tile (kt,nt):
//   k = 32*kt + (e&7) + 16*(e>=8) + 8*(L>=16),  n = 16*nt + (L&15)
__global__ void pack_b(const float* __restrict__ W, bf16_t* __restrict__ Bp, int K, int Nc) {
    int idx = blockIdx.x * blockDim.x + threadIdx.x;
    int total = K * Nc;
    if (idx >= total) return;
    int e = idx & 15;
    int t = idx >> 4;
    int lane = t & 31; t >>= 5;
    int ntiles = Nc >> 4;
    int nt = t % ntiles, kt = t / ntiles;
    int k = kt * 32 + (e & 7) + ((e >> 3) << 4) + ((lane >> 4) << 3);
    int n = nt * 16 + (lane & 15);
    Bp[idx] = (bf16_t)W[(long)k * Nc + n];
}

// ---------- WMMA GEMM (M-blocked 64x16 per wave) ----------
// C[M,Nc] = A[M,K](bf16,row-major) x Bp(packed bf16), fp32 acc.
// Each wave owns a 64-row x 16-col block: 4 accumulators share one B fragment
// per K step (4x fewer B fetches, 4 WMMAs per 5 b128-load pairs).
// Partial M blocks: compute on in-bounds staging garbage, guard only stores
// (wave-uniform guards keep EXEC all-1s for WMMA).
// optional epilogue: +bias[n], eval-BN, ReLU
__global__ void gemm_wmma(const bf16_t* __restrict__ A, const bf16_t* __restrict__ Bp,
                          float* __restrict__ Cc, int M, int K, int Nc,
                          const float* __restrict__ bias,
                          const float* __restrict__ g, const float* __restrict__ bb,
                          const float* __restrict__ mn, const float* __restrict__ vr,
                          int relu) {
    const int lane = threadIdx.x & 31;
    const int wib  = threadIdx.x >> 5;
    const int wpb  = blockDim.x >> 5;
    const int ntiles  = Nc >> 4;
    const int mblocks = (M + 63) >> 6;
    int wtile = blockIdx.x * wpb + wib;
    if (wtile >= mblocks * ntiles) return;   // wave-uniform exit
    const int mb = wtile / ntiles, nt = wtile % ntiles;
    const int m0 = mb << 6;
    const int hi = lane >> 4;                // 0: k 0-7,16-23; 1: k 8-15,24-31
    const bf16_t* arow0 = A + (long)(m0 + (lane & 15)) * K + hi * 8;
    const long rstride = (long)16 * K;       // 16 rows between subtiles
    v8f acc0 = {}, acc1 = {}, acc2 = {}, acc3 = {};
    const int ktc = K >> 5;
    for (int kt = 0; kt < ktc; ++kt) {
        const bf16_t* ap = arow0 + kt * 32;
        __builtin_prefetch(ap + 32, 0, 0);   // global_prefetch_b8 next K tile
        v16bf b = *(const v16bf*)(Bp + (((long)(kt * ntiles + nt) * 32 + lane) << 4));
        {
            v8bf alo = *(const v8bf*)(ap);
            v8bf ahi = *(const v8bf*)(ap + 16);
            v16bf a = __builtin_shufflevector(alo, ahi, 0,1,2,3,4,5,6,7,8,9,10,11,12,13,14,15);
            acc0 = __builtin_amdgcn_wmma_f32_16x16x32_bf16(false, a, false, b, (short)0, acc0,
                                                           false, false);
        }
        {
            v8bf alo = *(const v8bf*)(ap + rstride);
            v8bf ahi = *(const v8bf*)(ap + rstride + 16);
            v16bf a = __builtin_shufflevector(alo, ahi, 0,1,2,3,4,5,6,7,8,9,10,11,12,13,14,15);
            acc1 = __builtin_amdgcn_wmma_f32_16x16x32_bf16(false, a, false, b, (short)0, acc1,
                                                           false, false);
        }
        {
            v8bf alo = *(const v8bf*)(ap + 2 * rstride);
            v8bf ahi = *(const v8bf*)(ap + 2 * rstride + 16);
            v16bf a = __builtin_shufflevector(alo, ahi, 0,1,2,3,4,5,6,7,8,9,10,11,12,13,14,15);
            acc2 = __builtin_amdgcn_wmma_f32_16x16x32_bf16(false, a, false, b, (short)0, acc2,
                                                           false, false);
        }
        {
            v8bf alo = *(const v8bf*)(ap + 3 * rstride);
            v8bf ahi = *(const v8bf*)(ap + 3 * rstride + 16);
            v16bf a = __builtin_shufflevector(alo, ahi, 0,1,2,3,4,5,6,7,8,9,10,11,12,13,14,15);
            acc3 = __builtin_amdgcn_wmma_f32_16x16x32_bf16(false, a, false, b, (short)0, acc3,
                                                           false, false);
        }
    }
    const int col = (nt << 4) + (lane & 15);
    float bi = bias ? bias[col] : 0.f;
    float sc = 1.f, sh = 0.f;
    if (g) {
        float inv = rsqrtf(vr[col] + 1e-5f);
        sc = g[col] * inv;
        sh = bb[col] - mn[col] * sc;
    }
    v8f accs[4] = { acc0, acc1, acc2, acc3 };
    for (int i = 0; i < 4; ++i) {
        int ms = m0 + (i << 4);
        if (ms >= M) break;                  // wave-uniform store guard
        for (int r = 0; r < 8; ++r) {
            int rr = ms + r + hi * 8;        // C/D layout: VGPR r -> M=r (+8 lanes 16..31)
            float v = accs[i][r] + bi;
            if (g) v = v * sc + sh;
            if (relu) v = fmaxf(v, 0.f);
            Cc[(long)rr * Nc + col] = v;
        }
    }
}

// ---------- GAT attention pieces ----------
__global__ void node_dots(const float* __restrict__ h, const float* __restrict__ as_,
                          const float* __restrict__ ad_, float* hs, float* hd, float* sl,
                          unsigned* mx, int n, int F) {
    int wave = (blockIdx.x * blockDim.x + threadIdx.x) >> 5;
    int lane = threadIdx.x & 31;
    if (wave >= n) return;
    const float* hp = h + (long)wave * F;
    float s0 = 0.f, s1 = 0.f;
    for (int f = lane; f < F; f += 32) { float v = hp[f]; s0 += v * as_[f]; s1 += v * ad_[f]; }
    s0 = wave_sum(s0); s1 = wave_sum(s1);
    if (lane == 0) {
        hs[wave] = s0; hd[wave] = s1;
        float l = s0 + s1; l = (l > 0.f) ? l : 0.2f * l;    // leaky_relu(0.2)
        sl[wave] = l;
        mx[wave] = f2ord(l);                                 // self-loop seeds segment max
    }
}
__global__ void edge_max(const int* __restrict__ src, const int* __restrict__ dst,
                         const int* __restrict__ msk, const float* __restrict__ hs,
                         const float* __restrict__ hd, float* logit, unsigned* mx, int ne) {
    int e = blockIdx.x * blockDim.x + threadIdx.x;
    if (e >= ne) return;
    if (!msk[e]) { logit[e] = -1e9f; return; }
    int s = src[e], d = dst[e];
    float l = hs[s] + hd[d]; l = (l > 0.f) ? l : 0.2f * l;
    logit[e] = l;
    atomicMax(mx + d, f2ord(l));
}
__global__ void node_selfexp(const float* sl, const unsigned* mx, float* selfw, float* den, int n) {
    int i = blockIdx.x * blockDim.x + threadIdx.x;
    if (i >= n) return;
    float w = __expf(sl[i] - ord2f(mx[i]));
    selfw[i] = w; den[i] = w;
}
__global__ void edge_exp(const int* __restrict__ dst, const int* __restrict__ msk,
                         const float* __restrict__ logit, const unsigned* __restrict__ mx,
                         float* w, float* den, int ne) {
    int e = blockIdx.x * blockDim.x + threadIdx.x;
    if (e >= ne) return;
    if (!msk[e]) { w[e] = 0.f; return; }
    int d = dst[e];
    float ww = __expf(logit[e] - ord2f(mx[d]));
    w[e] = ww;
    atomicAdd(den + d, ww);
}
__global__ void node_base(const float* __restrict__ h, const float* __restrict__ selfw,
                          const float* __restrict__ den, float* out, long n, int F) {
    long i = (long)blockIdx.x * blockDim.x + threadIdx.x;
    if (i >= n * F) return;
    long node = i / F;
    out[i] = (selfw[node] / den[node]) * h[i];
}
__global__ void edge_aggr(const int* __restrict__ src, const int* __restrict__ dst,
                          const int* __restrict__ msk, const float* __restrict__ w,
                          const float* __restrict__ den, const float* __restrict__ h,
                          float* out, int ne, int F) {
    int wave = (blockIdx.x * blockDim.x + threadIdx.x) >> 5;
    int lane = threadIdx.x & 31;
    if (wave >= ne) return;
    if (!msk[wave]) return;
    int s = src[wave], d = dst[wave];
    float alpha = w[wave] / den[d];
    const float* hp = h + (long)s * F;
    float* op = out + (long)d * F;
    for (int f = lane; f < F; f += 32) atomicAdd(op + f, alpha * hp[f]);
}
__global__ void bias_bn_relu(float* x, const float* __restrict__ bias, const float* __restrict__ g,
                             const float* __restrict__ b, const float* __restrict__ m,
                             const float* __restrict__ v, long n, int F) {
    long i = (long)blockIdx.x * blockDim.x + threadIdx.x;
    if (i >= n) return;
    int f = (int)(i % F);
    float val = x[i] + bias[f];
    val = (val - m[f]) * rsqrtf(v[f] + 1e-5f) * g[f] + b[f];
    x[i] = fmaxf(val, 0.f);
}

// ---------- TopK pooling ----------
__global__ void pnorm_k(const float* __restrict__ p, float* out, int F) {
    __shared__ float s[256];
    float v = (threadIdx.x < (unsigned)F) ? p[threadIdx.x] : 0.f;
    s[threadIdx.x] = v * v;
    __syncthreads();
    for (int o = 128; o > 0; o >>= 1) {
        if ((int)threadIdx.x < o) s[threadIdx.x] += s[threadIdx.x + o];
        __syncthreads();
    }
    if (threadIdx.x == 0) out[0] = sqrtf(s[0]) + 1e-16f;
}
__global__ void scores_k(const float* __restrict__ x, const float* __restrict__ p,
                         const float* __restrict__ nrm, float* sc, int n, int F) {
    int wave = (blockIdx.x * blockDim.x + threadIdx.x) >> 5;
    int lane = threadIdx.x & 31;
    if (wave >= n) return;
    const float* xp = x + (long)wave * F;
    float s = 0.f;
    for (int f = lane; f < F; f += 32) s += xp[f] * p[f];
    s = wave_sum(s);
    if (lane == 0) sc[wave] = s / nrm[0];
}
// one block (512 thr) per graph: bitonic sort 1024 (val desc, idx asc tie-break)
__global__ void topk_sort(const float* __restrict__ sc, int n_per, int k,
                          int* gidx, float* tanhv, int* remap) {
    __shared__ float sval[1024];
    __shared__ int   sidx[1024];
    int g = blockIdx.x;
    for (int i = threadIdx.x; i < 1024; i += blockDim.x) {
        if (i < n_per) { sval[i] = sc[g * n_per + i]; sidx[i] = i; }
        else           { sval[i] = -3.4e38f;          sidx[i] = 0x7fffffff; }
    }
    for (int kk = 2; kk <= 1024; kk <<= 1) {
        for (int j = kk >> 1; j > 0; j >>= 1) {
            __syncthreads();
            for (int i = threadIdx.x; i < 1024; i += blockDim.x) {
                int l = i ^ j;
                if (l > i) {
                    bool desc = ((i & kk) == 0);
                    float vi = sval[i], vl = sval[l];
                    int   ii = sidx[i], il = sidx[l];
                    bool lPrec = (vl > vi) || (vl == vi && il < ii);
                    bool iPrec = (vi > vl) || (vi == vl && ii < il);
                    if (desc ? lPrec : iPrec) {
                        sval[i] = vl; sval[l] = vi; sidx[i] = il; sidx[l] = ii;
                    }
                }
            }
        }
    }
    __syncthreads();
    for (int t = threadIdx.x; t < k; t += blockDim.x) {
        int oldG = g * n_per + sidx[t];
        int newG = g * k + t;
        gidx[newG]  = oldG;
        tanhv[newG] = tanhf(sval[t]);
        remap[oldG] = newG;
    }
}
__global__ void gather_gate(const float* __restrict__ xin, const int* __restrict__ gidx,
                            const float* __restrict__ tanhv, float* xout, long m, int F) {
    long i = (long)blockIdx.x * blockDim.x + threadIdx.x;
    if (i >= m * F) return;
    long node = i / F, f = i % F;
    xout[i] = xin[(long)gidx[node] * F + f] * tanhv[node];
}
__global__ void edge_remap(const int* __restrict__ src, const int* __restrict__ dst,
                           const int* __restrict__ msk, const int* __restrict__ remap,
                           int* ns, int* nd, int* nm, int ne) {
    int e = blockIdx.x * blockDim.x + threadIdx.x;
    if (e >= ne) return;
    int m = msk[e], s = 0, d = 0;
    if (m) {
        s = remap[src[e]]; d = remap[dst[e]];
        if (s < 0 || d < 0) { m = 0; s = 0; d = 0; }
    }
    ns[e] = s; nd[e] = d; nm[e] = m;
}
// readout accum: xsum[g, 0:F]=+max, xsum[g, F:2F]=+mean
__global__ void readout(const float* __restrict__ x, int k, float* xsum, int Bn, int F) {
    int t = blockIdx.x * blockDim.x + threadIdx.x;
    if (t >= Bn * F) return;
    int g = t / F, f = t % F;
    const float* xp = x + ((long)g * k) * F + f;
    float mx = -3.4e38f, sm = 0.f;
    for (int i = 0; i < k; ++i) { float v = xp[(long)i * F]; mx = fmaxf(mx, v); sm += v; }
    xsum[g * (2 * F) + f]     += mx;
    xsum[g * (2 * F) + F + f] += sm / (float)k;
}
__global__ void final_fc(const float* __restrict__ y, const float* __restrict__ W,
                         const float* __restrict__ b, float* out, int Bn, int K, int Cc) {
    int t = blockIdx.x * blockDim.x + threadIdx.x;
    if (t >= Bn * Cc) return;
    int g = t / Cc, c = t % Cc;
    float s = b[c];
    for (int k = 0; k < K; ++k) s += y[g * K + k] * W[k * Cc + c];
    out[t] = s;
}

// =====================================================================
extern "C" void kernel_launch(void* const* d_in, const int* in_sizes, int n_in,
                              void* d_out, int out_size, void* d_ws, size_t ws_size,
                              hipStream_t stream) {
    (void)in_sizes; (void)n_in; (void)out_size; (void)ws_size;
    const float* x0   = (const float*)d_in[0];
    const int*   es0  = (const int*)d_in[1];
    const int*   ed0  = (const int*)d_in[2];
    const float* W[3]   = { (const float*)d_in[3],  (const float*)d_in[8],  (const float*)d_in[13] };
    const float* as_[3] = { (const float*)d_in[4],  (const float*)d_in[9],  (const float*)d_in[14] };
    const float* ad_[3] = { (const float*)d_in[5],  (const float*)d_in[10], (const float*)d_in[15] };
    const float* bc_[3] = { (const float*)d_in[6],  (const float*)d_in[11], (const float*)d_in[16] };
    const float* pv_[3] = { (const float*)d_in[7],  (const float*)d_in[12], (const float*)d_in[17] };
    const float* bn[6][4];
    for (int i = 0; i < 6; ++i)
        for (int j = 0; j < 4; ++j) bn[i][j] = (const float*)d_in[18 + i * 4 + j];
    const float* lw1 = (const float*)d_in[42]; const float* lb1 = (const float*)d_in[43];
    const float* lw2 = (const float*)d_in[44]; const float* lb2 = (const float*)d_in[45];
    const float* lw3 = (const float*)d_in[46]; const float* lb3 = (const float*)d_in[47];

    // ---- workspace carve ----
    char* cur = (char*)d_ws;
    auto alloc = [&](size_t bytes) -> void* {
        void* p = (void*)cur; cur += (bytes + 255) & ~(size_t)255; return p;
    };
    bf16_t*   ABF  = (bf16_t*)  alloc((size_t)14 * 1024 * 1024);          // bf16 A staging
    bf16_t*   WP   = (bf16_t*)  alloc(512 * 256 * sizeof(bf16_t));        // packed B fragments
    float*    Hh   = (float*)   alloc((size_t)NODES0 * HDIM * 4);         // h = A@W
    float*    OUT  = (float*)   alloc((size_t)NODES0 * HDIM * 4);         // gat output / pooled input
    float*    X1   = (float*)   alloc((size_t)B_G * KP1 * HDIM * 4);      // gathered x
    float*    HS   = (float*)   alloc(NODES0 * 4);
    float*    HD   = (float*)   alloc(NODES0 * 4);
    float*    SL   = (float*)   alloc(NODES0 * 4);
    float*    SW   = (float*)   alloc(NODES0 * 4);
    float*    DEN  = (float*)   alloc(NODES0 * 4);
    unsigned* MX   = (unsigned*)alloc(NODES0 * 4);
    float*    LOG_ = (float*)   alloc((size_t)NE * 4);
    float*    WE   = (float*)   alloc((size_t)NE * 4);
    int* eAs = (int*)alloc(NE * 4); int* eAd = (int*)alloc(NE * 4); int* eAm = (int*)alloc(NE * 4);
    int* eBs = (int*)alloc(NE * 4); int* eBd = (int*)alloc(NE * 4); int* eBm = (int*)alloc(NE * 4);
    float* SC    = (float*)alloc(NODES0 * 4);
    float* PN    = (float*)alloc(256);
    int*   GIDX  = (int*)  alloc(B_G * KP1 * 4);
    float* TANHV = (float*)alloc(B_G * KP1 * 4);
    int*   REMAP = (int*)  alloc(NODES0 * 4);
    float* XSUM  = (float*)alloc(B_G * 2 * HDIM * 4);
    float* Y1    = (float*)alloc(B_G * HDIM * 4);
    float* Y2    = (float*)alloc(B_G * 128 * 4);

    auto cdiv = [](long a, long b) { return (int)((a + b - 1) / b); };

    auto run_gemm = [&](const bf16_t* A, const float* Wsrc, float* C, int M, int K, int Nc,
                        const float* bias, const float* g, const float* bb,
                        const float* mn, const float* vr, int relu) {
        pack_b<<<cdiv((long)K * Nc, 256), 256, 0, stream>>>(Wsrc, WP, K, Nc);
        int tiles = ((M + 63) / 64) * (Nc / 16);
        gemm_wmma<<<cdiv(tiles, 4), 128, 0, stream>>>(A, WP, C, M, K, Nc,
                                                      bias, g, bb, mn, vr, relu);
    };

    auto gat_layer = [&](const float* xin, int Fin, int n,
                         int li /*0..2*/, int bnpost /*bn idx*/, int bnin /*-1 or bn idx*/,
                         const int* s, const int* d, const int* m) {
        long nel = (long)n * Fin;
        const float* ig = (bnin >= 0) ? bn[bnin][0] : nullptr;
        const float* ib = (bnin >= 0) ? bn[bnin][1] : nullptr;
        const float* im = (bnin >= 0) ? bn[bnin][2] : nullptr;
        const float* iv = (bnin >= 0) ? bn[bnin][3] : nullptr;
        cvt_bf16<<<cdiv(nel, 256), 256, 0, stream>>>(xin, ABF, nel, Fin, ig, ib, im, iv);
        run_gemm(ABF, W[li], Hh, n, Fin, HDIM, nullptr, nullptr, nullptr, nullptr, nullptr, 0);
        node_dots<<<cdiv(n, 8), 256, 0, stream>>>(Hh, as_[li], ad_[li], HS, HD, SL, MX, n, HDIM);
        edge_max<<<cdiv(NE, 256), 256, 0, stream>>>(s, d, m, HS, HD, LOG_, MX, NE);
        node_selfexp<<<cdiv(n, 256), 256, 0, stream>>>(SL, MX, SW, DEN, n);
        edge_exp<<<cdiv(NE, 256), 256, 0, stream>>>(d, m, LOG_, MX, WE, DEN, NE);
        node_base<<<cdiv((long)n * HDIM, 256), 256, 0, stream>>>(Hh, SW, DEN, OUT, n, HDIM);
        edge_aggr<<<cdiv(NE, 8), 256, 0, stream>>>(s, d, m, WE, DEN, Hh, OUT, NE, HDIM);
        bias_bn_relu<<<cdiv((long)n * HDIM, 256), 256, 0, stream>>>(
            OUT, bc_[li], bn[bnpost][0], bn[bnpost][1], bn[bnpost][2], bn[bnpost][3],
            (long)n * HDIM, HDIM);
    };

    auto topk_stage = [&](int n_per, int k, int li,
                          const int* si, const int* di, const int* mi,
                          int* so, int* dd, int* mo, bool do_edges) {
        int n = B_G * n_per;
        pnorm_k<<<1, 256, 0, stream>>>(pv_[li], PN, HDIM);
        scores_k<<<cdiv(n, 8), 256, 0, stream>>>(OUT, pv_[li], PN, SC, n, HDIM);
        fill_i32<<<cdiv(n, 256), 256, 0, stream>>>(REMAP, -1, n);
        topk_sort<<<B_G, 512, 0, stream>>>(SC, n_per, k, GIDX, TANHV, REMAP);
        gather_gate<<<cdiv((long)B_G * k * HDIM, 256), 256, 0, stream>>>(
            OUT, GIDX, TANHV, X1, (long)B_G * k, HDIM);
        if (do_edges)
            edge_remap<<<cdiv(NE, 256), 256, 0, stream>>>(si, di, mi, REMAP, so, dd, mo, NE);
        readout<<<cdiv(B_G * HDIM, 256), 256, 0, stream>>>(X1, k, XSUM, B_G, HDIM);
    };

    // ---- pipeline ----
    fill_f32<<<cdiv(B_G * 2 * HDIM, 256), 256, 0, stream>>>(XSUM, 0.f, B_G * 2 * HDIM);
    init_edges<<<cdiv(NE, 256), 256, 0, stream>>>(es0, ed0, eAs, eAd, eAm, NE);

    // layer 1: input bn1, post bn2
    gat_layer(x0, F_IN, NODES0, 0, /*bnpost=*/1, /*bnin=*/0, eAs, eAd, eAm);
    topk_stage(N0, KP1, 0, eAs, eAd, eAm, eBs, eBd, eBm, true);
    // layer 2: post bn3
    gat_layer(X1, HDIM, B_G * KP1, 1, 2, -1, eBs, eBd, eBm);
    topk_stage(KP1, KP2, 1, eBs, eBd, eBm, eAs, eAd, eAm, true);
    // layer 3: post bn4
    gat_layer(X1, HDIM, B_G * KP2, 2, 3, -1, eAs, eAd, eAm);
    topk_stage(KP2, KP3, 2, eAs, eAd, eAm, eBs, eBd, eBm, false);

    // ---- MLP head ----
    cvt_bf16<<<cdiv((long)B_G * 512, 256), 256, 0, stream>>>(
        XSUM, ABF, (long)B_G * 512, 512, nullptr, nullptr, nullptr, nullptr);
    run_gemm(ABF, lw1, Y1, B_G, 512, HDIM, lb1, bn[4][0], bn[4][1], bn[4][2], bn[4][3], 1);
    cvt_bf16<<<cdiv((long)B_G * HDIM, 256), 256, 0, stream>>>(
        Y1, ABF, (long)B_G * HDIM, HDIM, nullptr, nullptr, nullptr, nullptr);
    run_gemm(ABF, lw2, Y2, B_G, HDIM, 128, lb2, bn[5][0], bn[5][1], bn[5][2], bn[5][3], 1);
    final_fc<<<cdiv(B_G * NCLS, 256), 256, 0, stream>>>(Y2, lw3, lb3, (float*)d_out,
                                                        B_G, 128, NCLS);
}
